// ContextualAttention_65077344469480
// MI455X (gfx1250) — matrix operand order repaired and checked by reading, not compiled, not run on hardware
//
#include <hip/hip_runtime.h>
#include <math.h>

// ---------------------------------------------------------------------------
// Contextual attention, restructured as:
//   build W[p][k], FPT[l][k], Vt[n][p]  (bf16)
//   GEMM1 (WMMA bf16):  S0[p][l]  = W · FPT^T        (2304 x 2304 x 1152)
//   fused diagonal 3-tap x2 (exact composition incl. flat wraparound)
//   softmax over p (mask pre+post), write At[l][p] bf16
//   GEMM2 (WMMA bf16):  OB[l][n]  = At · Vt^T        (2304 x 2048 x 2304)
//   gather overlapping 4x4 blocks -> out (96x96), * 0.25
// ---------------------------------------------------------------------------

typedef __bf16 bf16_t;
typedef __attribute__((ext_vector_type(16))) __bf16 bf16x16;
typedef __attribute__((ext_vector_type(8)))  __bf16 bf16x8;
typedef __attribute__((ext_vector_type(8)))  float  f32x8;
typedef int v4i_vs __attribute__((vector_size(16)));   // matches builtin param

static constexpr int kC    = 128;
static constexpr int kFull = 96;
static constexpr int kDs   = 48;            // downsampled H/W
static constexpr int kP    = kDs * kDs;     // 2304 patches / locations
static constexpr int kK1   = kC * 9;        // 1152 features (C*3*3)
static constexpr int kN2   = kC * 16;       // 2048 (C * 4*4 raw patch)
static constexpr int kB    = 8;

// ---- CDNA5 async global->LDS staging (probe with __has_builtin) ------------
#if defined(__has_builtin)
#  if __has_builtin(__builtin_amdgcn_global_load_async_to_lds_b128)
#    define ASYNC_LDS 1
#  else
#    define ASYNC_LDS 0
#  endif
#  if __has_builtin(__builtin_amdgcn_s_wait_asynccnt)
#    define WAIT_ASYNC_BUILTIN 1
#  else
#    define WAIT_ASYNC_BUILTIN 0
#  endif
#else
#  define ASYNC_LDS 0
#  define WAIT_ASYNC_BUILTIN 0
#endif

__device__ __forceinline__ void stage16(const bf16_t* __restrict__ g, bf16_t* l)
{
#if ASYNC_LDS
    __builtin_amdgcn_global_load_async_to_lds_b128(
        (__attribute__((address_space(1))) v4i_vs*)g,
        (__attribute__((address_space(3))) v4i_vs*)l, 0, 0);
#else
    *(bf16x8*)l = *(const bf16x8*)g;
#endif
}

__device__ __forceinline__ void wait_async()
{
#if ASYNC_LDS
#  if WAIT_ASYNC_BUILTIN
    __builtin_amdgcn_s_wait_asynccnt(0);
#  else
    asm volatile("s_wait_asynccnt 0x0" ::: "memory");
#  endif
#endif
}

// ---------------- mask validity per background patch -----------------------
__global__ void mm_kernel(const float* __restrict__ mask, float* __restrict__ mm)
{
    int p = blockIdx.x * blockDim.x + threadIdx.x;
    if (p >= kP) return;
    int pi = p / kDs, pj = p % kDs;
    float s = 0.f;
    for (int i = 0; i < 3; ++i)
        for (int j = 0; j < 3; ++j) {
            int y = pi - 1 + i, x = pj - 1 + j;
            if (y >= 0 && y < kDs && x >= 0 && x < kDs)
                s += mask[(2 * y) * kFull + 2 * x];
        }
    mm[p] = ((s * (1.0f / 9.0f)) == 0.0f) ? 1.0f : 0.0f;
}

// ---------------- normalized background patches W[p][k] ---------------------
__global__ void __launch_bounds__(128)
build_w_kernel(const float* __restrict__ bsrc, bf16_t* __restrict__ W, int batch)
{
    __shared__ float red[128];
    const int p   = blockIdx.x;
    const int pi  = p / kDs, pj = p % kDs;
    const int tid = threadIdx.x;
    float vals[9];
    float ss = 0.f;
#pragma unroll
    for (int it = 0; it < 9; ++it) {
        int k = tid + it * 128;                // covers 0..1151
        int c = k / 9, rem = k % 9;
        int i = rem / 3, j = rem % 3;
        int y = pi - 1 + i, x = pj - 1 + j;
        float v = 0.f;
        if (y >= 0 && y < kDs && x >= 0 && x < kDs)
            v = bsrc[(((size_t)batch * kC + c) * kFull + 2 * y) * kFull + 2 * x];
        vals[it] = v;
        ss += v * v;
    }
    red[tid] = ss;
    __syncthreads();
    for (int s = 64; s > 0; s >>= 1) {
        if (tid < s) red[tid] += red[tid + s];
        __syncthreads();
    }
    // sum(bp*bp + 1e-4) adds 1e-4 for every one of the 1152 elements
    float norm = sqrtf(red[0] + (float)kK1 * 1.0e-4f);
#pragma unroll
    for (int it = 0; it < 9; ++it) {
        int k = tid + it * 128;
        W[(size_t)p * kK1 + k] = (bf16_t)(vals[it] / norm);
    }
}

// ---------------- foreground patches FPT[l][k] ------------------------------
__global__ void build_fp_kernel(const float* __restrict__ f, bf16_t* __restrict__ FPT, int batch)
{
    int id = blockIdx.x * blockDim.x + threadIdx.x;
    if (id >= kP * kK1) return;
    int l = id / kK1, k = id % kK1;
    int c = k / 9, rem = k % 9, i = rem / 3, j = rem % 3;
    int li = l / kDs, lj = l % kDs;
    int y = li - 1 + i, x = lj - 1 + j;
    float v = 0.f;
    if (y >= 0 && y < kDs && x >= 0 && x < kDs)
        v = f[(((size_t)batch * kC + c) * kFull + 2 * y) * kFull + 2 * x];
    FPT[id] = (bf16_t)v;
}

// ---------------- raw 4x4 background patches Vt[n][p] -----------------------
__global__ void build_v_kernel(const float* __restrict__ bsrc, bf16_t* __restrict__ Vt, int batch)
{
    int id = blockIdx.x * blockDim.x + threadIdx.x;
    if (id >= kN2 * kP) return;
    int n = id / kP, p = id % kP;          // p fastest -> coalesced writes
    int c = n >> 4, kh = (n >> 2) & 3, kw = n & 3;
    int ph = p / kDs, pw = p % kDs;
    int y = 2 * ph - 1 + kh, x = 2 * pw - 1 + kw;   // full-res, pad 1
    float v = 0.f;
    if (y >= 0 && y < kFull && x >= 0 && x < kFull)
        v = bsrc[(((size_t)batch * kC + c) * kFull + y) * kFull + x];
    Vt[id] = (bf16_t)v;
}

// ---------------- bf16 NT GEMM via v_wmma_f32_16x16x32_bf16 -----------------
// C[m][n] = sum_k A[m][k] * Bt[n][k].  M % 128 == 0, N % 256 == 0, K % 32 == 0.
// Block tile 128x256, 8 wave32s in a 2x4 grid, 64x64 per wave (16 WMMAs/stage).
__global__ void __launch_bounds__(256)
gemm_bf16_nt(const bf16_t* __restrict__ A, const bf16_t* __restrict__ Bt,
             float* __restrict__ Cm, int M, int N, int K)
{
    constexpr int BM = 128, BN = 256, BK = 32, PAD = 8;
    __shared__ bf16_t As[BM][BK + PAD];
    __shared__ bf16_t Bs[BN][BK + PAD];

    const int tid  = threadIdx.x;
    const int lane = tid & 31;
    const int wave = tid >> 5;      // 8 wave32s
    const int wm   = wave & 1;      // 2 wave rows (M), 64 each
    const int wn   = wave >> 1;     // 4 wave cols (N), 64 each
    const int half = lane >> 4;     // K-half select
    const int lr   = lane & 15;

    const int m0 = blockIdx.y * BM;
    const int n0 = blockIdx.x * BN;

    f32x8 acc[4][4] = {};

    for (int k0 = 0; k0 < K; k0 += BK) {
        // stage A:128x32 (512 x 16B) and B:256x32 (1024 x 16B); 6 chunks/thread
#pragma unroll
        for (int i = 0; i < 6; ++i) {
            int chunk = tid + i * 256;               // 0..1535
            if (chunk < 512) {
                int row = chunk >> 2;
                int col = (chunk & 3) * 8;
                stage16(&A[(size_t)(m0 + row) * K + k0 + col], &As[row][col]);
            } else {
                int ch  = chunk - 512;
                int row = ch >> 2;
                int col = (ch & 3) * 8;
                stage16(&Bt[(size_t)(n0 + row) * K + k0 + col], &Bs[row][col]);
            }
        }
        wait_async();
        __syncthreads();

        // A fragment: row = lane%16, K = {half*8..+8} U {16+half*8..+8}
        bf16x16 afr[4];
#pragma unroll
        for (int mt = 0; mt < 4; ++mt) {
            const bf16_t* p = &As[wm * 64 + mt * 16 + lr][0];
            bf16x8 lo = *(const bf16x8*)(p + half * 8);
            bf16x8 hi = *(const bf16x8*)(p + 16 + half * 8);
            afr[mt] = __builtin_shufflevector(lo, hi, 0,1,2,3,4,5,6,7,8,9,10,11,12,13,14,15);
        }
        // B fragment (column-major): col = lane%16, K = {half*8..+8} U {16+half*8..+8}
        bf16x16 bfr[4];
#pragma unroll
        for (int nt = 0; nt < 4; ++nt) {
            const bf16_t* p = &Bs[wn * 64 + nt * 16 + lr][0];
            bf16x8 lo = *(const bf16x8*)(p + half * 8);
            bf16x8 hi = *(const bf16x8*)(p + 16 + half * 8);
            bfr[nt] = __builtin_shufflevector(lo, hi, 0,1,2,3,4,5,6,7,8,9,10,11,12,13,14,15);
        }
#pragma unroll
        for (int mt = 0; mt < 4; ++mt)
#pragma unroll
            for (int nt = 0; nt < 4; ++nt)
                acc[mt][nt] = __builtin_amdgcn_wmma_f32_16x16x32_bf16(
                    false, afr[mt], false, bfr[nt], (short)0, acc[mt][nt], false, false);
        __syncthreads();
    }

    // D layout: VGPR r -> row r + 8*(lane/16); col = lane%16.
    // Walk a row pointer to keep the address math additive.
    const int colbase = n0 + wn * 64 + lr;
#pragma unroll
    for (int mt = 0; mt < 4; ++mt) {
        float* prow = Cm + (size_t)(m0 + wm * 64 + mt * 16 + 8 * half) * N + colbase;
#pragma unroll
        for (int r = 0; r < 8; ++r) {
#pragma unroll
            for (int nt = 0; nt < 4; ++nt)
                prow[nt * 16] = acc[mt][nt][r];
            prow += N;
        }
    }
}

// ---------------- fused double diagonal 3-tap (exact composition) -----------
__device__ __forceinline__ int orig_flat(int X) { return (X % kDs) * kDs + (X / kDs); }

__global__ void fuse_kernel(const float* __restrict__ S0, float* __restrict__ S1)
{
    int id = blockIdx.x * blockDim.x + threadIdx.x;
    if (id >= kP * kP) return;
    int Pp = id / kP, Lp = id % kP;          // transposed flat coords (stage 2)
    float acc = 0.f;
#pragma unroll
    for (int d2 = -1; d2 <= 1; ++d2) {
        int Pd = Pp + d2, Ld = Lp + d2;
        if (Pd < 0 || Pd >= kP || Ld < 0 || Ld >= kP) continue;
        int P = orig_flat(Pd), L = orig_flat(Ld);
#pragma unroll
        for (int d1 = -1; d1 <= 1; ++d1) {   // stage 1, original flat coords
            int P1 = P + d1, L1 = L + d1;
            if (P1 >= 0 && P1 < kP && L1 >= 0 && L1 < kP)
                acc += S0[(size_t)P1 * kP + L1];
        }
    }
    S1[(size_t)orig_flat(Pp) * kP + orig_flat(Lp)] = acc;
}

// ---------------- masked softmax over p; write At[l][p] bf16 ----------------
__global__ void __launch_bounds__(256)
softmax_kernel(const float* __restrict__ S, const float* __restrict__ mm,
               bf16_t* __restrict__ At)
{
    __shared__ float red[256];
    const int l   = blockIdx.x;
    const int tid = threadIdx.x;
    float x[9], mv[9];
    float lmax = -1e30f;
#pragma unroll
    for (int it = 0; it < 9; ++it) {
        int p = tid + it * 256;               // 9*256 == 2304
        float m = mm[p];
        mv[it] = m;
        float v = S[(size_t)p * kP + l] * m * 10.0f;   // SCALE = 10
        x[it] = v;
        lmax = fmaxf(lmax, v);
    }
    red[tid] = lmax; __syncthreads();
    for (int s = 128; s > 0; s >>= 1) { if (tid < s) red[tid] = fmaxf(red[tid], red[tid + s]); __syncthreads(); }
    float gmax = red[0]; __syncthreads();

    float lsum = 0.f;
#pragma unroll
    for (int it = 0; it < 9; ++it) { x[it] = expf(x[it] - gmax); lsum += x[it]; }
    red[tid] = lsum; __syncthreads();
    for (int s = 128; s > 0; s >>= 1) { if (tid < s) red[tid] += red[tid + s]; __syncthreads(); }
    float inv = 1.0f / red[0];

#pragma unroll
    for (int it = 0; it < 9; ++it) {
        int p = tid + it * 256;
        At[(size_t)l * kP + p] = (bf16_t)(x[it] * inv * mv[it]);
    }
}

// ---------------- gather overlapping 4x4 blocks into full-res output --------
__global__ void scatter_kernel(const float* __restrict__ OB, float* __restrict__ out, int batch)
{
    int id = blockIdx.x * blockDim.x + threadIdx.x;
    if (id >= kC * kFull * kFull) return;
    int c = id / (kFull * kFull);
    int h = (id / kFull) % kFull;
    int w = id % kFull;
    float acc = 0.f;
    int hc = h >> 1, wc = w >> 1;
    for (int dh = -1; dh <= 1; ++dh) {
        int hf = hc + dh;
        if (hf < 0 || hf >= kDs) continue;
        int kh = h + 1 - 2 * hf;
        if (kh < 0 || kh > 3) continue;
        for (int dw = -1; dw <= 1; ++dw) {
            int wf = wc + dw;
            if (wf < 0 || wf >= kDs) continue;
            int kw = w + 1 - 2 * wf;
            if (kw < 0 || kw > 3) continue;
            acc += OB[(size_t)(hf * kDs + wf) * kN2 + c * 16 + kh * 4 + kw];
        }
    }
    out[(((size_t)batch * kC + c) * kFull + h) * kFull + w] = acc * 0.25f;
}

// ---------------------------------------------------------------------------
extern "C" void kernel_launch(void* const* d_in, const int* in_sizes, int n_in,
                              void* d_out, int out_size, void* d_ws, size_t ws_size,
                              hipStream_t stream)
{
    (void)in_sizes; (void)n_in; (void)out_size; (void)ws_size;

    const float* f    = (const float*)d_in[0];
    const float* bsrc = (const float*)d_in[1];
    const float* mask = (const float*)d_in[2];
    float* out = (float*)d_out;

    char* ws = (char*)d_ws;
    size_t off = 0;
    auto take = [&](size_t bytes) -> char* {
        char* p = ws + off;
        off += (bytes + 255) & ~(size_t)255;
        return p;
    };

    float*  mmb = (float*) take((size_t)kP * 4);
    bf16_t* W   = (bf16_t*)take((size_t)kP * kK1 * 2);
    bf16_t* FPT = (bf16_t*)take((size_t)kP * kK1 * 2);
    bf16_t* Vt  = (bf16_t*)take((size_t)kN2 * kP * 2);
    float*  S0  = (float*) take((size_t)kP * kP * 4);
    float*  S1  = (float*) take((size_t)kP * kP * 4);
    bf16_t* At  = (bf16_t*)take((size_t)kP * kP * 2);
    float*  OB  = (float*) take((size_t)kP * kN2 * 4);

    mm_kernel<<<(kP + 255) / 256, 256, 0, stream>>>(mask, mmb);

    for (int batch = 0; batch < kB; ++batch) {
        build_w_kernel<<<kP, 128, 0, stream>>>(bsrc, W, batch);
        build_fp_kernel<<<(kP * kK1 + 255) / 256, 256, 0, stream>>>(f, FPT, batch);
        build_v_kernel<<<(kN2 * kP + 255) / 256, 256, 0, stream>>>(bsrc, Vt, batch);

        gemm_bf16_nt<<<dim3(kP / 256, kP / 128), 256, 0, stream>>>(W, FPT, S0, kP, kP, kK1);

        fuse_kernel<<<(kP * kP + 255) / 256, 256, 0, stream>>>(S0, S1);

        softmax_kernel<<<kP, 256, 0, stream>>>(S1, mmb, At);

        gemm_bf16_nt<<<dim3(kN2 / 256, kP / 128), 256, 0, stream>>>(At, Vt, OB, kP, kN2, kP);

        scatter_kernel<<<(kC * kFull * kFull + 255) / 256, 256, 0, stream>>>(OB, out, batch);
    }
}
